// SpatialEmbLoss_3d_74191265071110
// MI455X (gfx1250) — matrix operand order repaired and checked by reading, not compiled, not run on hardware
//
#include <hip/hip_runtime.h>
#include <hip/hip_bf16.h>

// ---- problem constants (from reference) ----
#define BB 4
#define CC 7
#define DD 32
#define HH 192
#define WW 192
#define DHW (DD*HH*WW)        // 1,179,648
#define NIDS 16
#define NBINS 2048
#define CHUNK1 8192           // voxels per block, phase 1
#define CHUNK3 16384          // voxels per block, phase 3

typedef __attribute__((ext_vector_type(16))) _Float16 v16h;
typedef __attribute__((ext_vector_type(8)))  float    v8f;
typedef __attribute__((ext_vector_type(4)))  unsigned v4u;
typedef __attribute__((ext_vector_type(8)))  int      v8i;
typedef __attribute__((ext_vector_type(4)))  int      v4i;

// ---- fast CDNA5 transcendentals ----
__device__ __forceinline__ float fast_tanh(float x) {
#if __has_builtin(__builtin_amdgcn_tanhf)
  return __builtin_amdgcn_tanhf(x);          // v_tanh_f32 (gfx1250)
#else
  float r;
  asm("v_tanh_f32 %0, %1\n\tv_nop\n\tv_nop" : "=v"(r) : "v"(x));
  return r;
#endif
}
__device__ __forceinline__ float fast_exp(float x) {
  return __expf(x);                          // v_exp_f32 path
}
__device__ __forceinline__ float fast_sigmoid(float x) {
  return __builtin_amdgcn_rcpf(1.f + __expf(-x));   // v_rcp_f32 + v_exp_f32
}

// =====================================================================
// Phase 1: per-instance stats via WMMA one-hot matmul + background seed.
//   stats[b][m][n], m=id-1 (0..15), n: 0..2 = sum xyzm, 3..5 = sum sigma,
//   6 = count.  bg[b] = sum seed^2 where label==0.
//   Instance-id tile is streamed into LDS with the Tensor Data Mover.
// =====================================================================
__global__ __launch_bounds__(256) void phase1_stats(
    const float* __restrict__ pred, const int* __restrict__ inst,
    const int* __restrict__ labels, const float* __restrict__ xyzm,
    float* __restrict__ stats, float* __restrict__ bg)
{
  __shared__ int instLds[CHUNK1];                  // 32 KB via TDM
  __shared__ _Float16 feat[8][32][16];             // per-wave B staging, 8 KB

  const int b   = blockIdx.y;
  const long long chunkBase = (long long)blockIdx.x * CHUNK1;
  const int tid = threadIdx.x;
  const int w   = tid >> 5;
  const int l   = tid & 31;

  // zero the feature staging (features 7..15 stay zero forever)
  for (int i = tid; i < 8*32*16; i += 256) ((_Float16*)feat)[i] = (_Float16)0.f;

  // ---- TDM: async DMA of this chunk's instance ids into LDS ----
  if (tid < 32) {
    unsigned long long ga =
        (unsigned long long)(const void*)(inst + (long long)b * DHW + chunkBase);
    unsigned ldsOff = (unsigned)(unsigned long long)(void*)&instLds[0];
    v4u g0; v8i g1; v4i g2 = {0,0,0,0}; v4i g3 = {0,0,0,0};
    v8i g4 = {0,0,0,0,0,0,0,0};
    // D# group0: count=1 | lds_addr | global_addr[56:0] | type=2
    g0[0] = 1u;
    g0[1] = ldsOff;
    g0[2] = (unsigned)(ga & 0xFFFFFFFFu);
    g0[3] = (unsigned)((ga >> 32) & 0x01FFFFFFu) | (2u << 30);
    // D# group1: data_size=4B, tensor_dim0=CHUNK1, tensor_dim1=1,
    //            tile_dim0=CHUNK1, tile_dim1=1, dim0_stride=CHUNK1
    g1[0] = (int)(2u << 16);                               // data_size = 4B
    g1[1] = (int)(((unsigned)CHUNK1 & 0xFFFFu) << 16);     // tensor_dim0 lo16
    g1[2] = (int)((((unsigned)CHUNK1 >> 16) & 0xFFFFu) | (1u << 16)); // hi16 | tensor_dim1 lo16=1
    g1[3] = (int)(((unsigned)CHUNK1) << 16);               // tensor_dim1 hi=0 | tile_dim0
    g1[4] = 1;                                             // tile_dim1=1, tile_dim2=0
    g1[5] = CHUNK1;                                        // tensor_dim0_stride lo32
    g1[6] = 0; g1[7] = 0;
    __builtin_amdgcn_tensor_load_to_lds(g0, g1, g2, g3, g4, 0);
    __builtin_amdgcn_s_wait_tensorcnt(0);
  }
  __syncthreads();

  const float* predB = pred + (long long)b * CC * DHW;
  v8f c = {0.f,0.f,0.f,0.f,0.f,0.f,0.f,0.f};
  float bgAcc = 0.f;
  const int n    = l & 15;                 // A row M / D column N
  const int hiA  = (l >= 16) ? 8 : 0;      // A-matrix K base per lane group
  const int hiB  = (l >= 16) ? 16 : 0;     // unused; B takes its K index from hiA, as A does
  const int myId = n + 1;

  for (int it = 0; it < CHUNK1/(32*8); ++it) {
    const int gb = (it*8 + w) * 32;              // 32-voxel group base (local)
    const long long v = chunkBase + gb + l;      // this lane's voxel (in-batch)

    float s0 = predB[3*(long long)DHW + v];
    float s1 = predB[4*(long long)DHW + v];
    float s2 = predB[5*(long long)DHW + v];
    float sd = fast_sigmoid(predB[6*(long long)DHW + v]);
    if (labels[(long long)b*DHW + v] == 0) bgAcc += sd * sd;
    float xm = xyzm[v], ym = xyzm[DHW + v], zm = xyzm[2*(long long)DHW + v];

    // stage this voxel's features (voxel-major) for the B matrix
    _Float16* fr = &feat[w][l][0];
    fr[0] = (_Float16)xm; fr[1] = (_Float16)ym; fr[2] = (_Float16)zm;
    fr[3] = (_Float16)s0; fr[4] = (_Float16)s1; fr[5] = (_Float16)s2;
    fr[6] = (_Float16)1.f;
    __builtin_amdgcn_wave_barrier();

    // A: one-hot(id) 16x32  (lane layout per ISA 7.12.2, 16-bit A 16x32)
    v16h a;
#pragma unroll
    for (int h = 0; h < 16; ++h) {
      int K = hiA + ((h < 8) ? h : (h + 8));
      a[h] = (instLds[gb + K] == myId) ? (_Float16)1.f : (_Float16)0.f;
    }
    // B: features 32x16 (K=voxel, N=feature)
    v16h bm;
#pragma unroll
    for (int h = 0; h < 16; ++h) bm[h] = feat[w][hiA + ((h < 8) ? h : (h + 8))][n];

    c = __builtin_amdgcn_wmma_f32_16x16x32_f16(false, a, false, bm,
                                               (short)0, c, false, false);
  }

  // flush D: lane holds (N = l&15) x (M = r + 8*(l>=16))
  const int mOfs = (l >= 16) ? 8 : 0;
  if (n < 7) {
#pragma unroll
    for (int r = 0; r < 8; ++r)
      atomicAdd(&stats[b*256 + (r + mOfs)*16 + n], c[r]);
  }
  // background seed: wave32 reduction
  for (int o = 16; o > 0; o >>= 1) bgAcc += __shfl_down(bgAcc, o, 32);
  if (l == 0) atomicAdd(&bg[b], bgAcc);
}

// =====================================================================
// Phase 2: per-(b,id) parameters: center, exp(10*s_mean), s_mean, cnt, exists
// =====================================================================
__global__ void phase2_params(const float* __restrict__ stats,
                              float* __restrict__ params)
{
  int t = threadIdx.x;
  if (t >= BB*NIDS) return;
  int b = t >> 4, id = t & 15;
  const float* s = stats + b*256 + id*16;
  float cnt  = s[6];
  float safe = fmaxf(cnt, 1.f);
  float* p = params + t*12;
  p[0] = s[0]/safe; p[1] = s[1]/safe; p[2] = s[2]/safe;          // center
#pragma unroll
  for (int f = 0; f < 3; ++f) {
    float sm = s[3+f]/safe;
    p[6+f] = sm;
    p[3+f] = expf(10.f * sm);                                    // s_exp
  }
  p[9]  = cnt;
  p[10] = (cnt > 0.f) ? 1.f : 0.f;
  p[11] = 0.f;
}

// =====================================================================
// Phase 3: per-(b,id) pass: dist field, error histograms (LDS->global),
//          seed loss and sigma variance accumulators.  L2-resident per b.
// =====================================================================
__global__ __launch_bounds__(256) void phase3_dist(
    const float* __restrict__ pred, const int* __restrict__ inst,
    const float* __restrict__ xyzm, const float* __restrict__ params,
    unsigned* __restrict__ hcp, unsigned* __restrict__ hcn,
    float* __restrict__ hsum, float* __restrict__ seedAcc,
    float* __restrict__ varAcc)
{
  __shared__ unsigned lcp[NBINS];
  __shared__ unsigned lcn[NBINS];
  __shared__ float    lsum[NBINS];

  const int b  = blockIdx.z;
  const int id = blockIdx.y;
  const int tid = threadIdx.x;
  for (int i = tid; i < NBINS; i += 256) { lcp[i]=0u; lcn[i]=0u; lsum[i]=0.f; }
  __syncthreads();

  const float* p = params + (b*NIDS + id)*12;
  const float cx=p[0], cy=p[1], cz=p[2];
  const float e0=p[3], e1=p[4], e2=p[5];
  const float sm0=p[6], sm1=p[7], sm2=p[8];
  const int iid = id + 1;
  const float* predB = pred + (long long)b * CC * DHW;
  const long long base = (long long)blockIdx.x * CHUNK3;

  float sA = 0.f, vA = 0.f;
  for (int it = 0; it < CHUNK3/256; ++it) {
    long long v = base + it*256 + tid;
    float se0 = fast_tanh(predB[v])                    + xyzm[v];
    float se1 = fast_tanh(predB[(long long)DHW   + v]) + xyzm[(long long)DHW   + v];
    float se2 = fast_tanh(predB[2*(long long)DHW + v]) + xyzm[2*(long long)DHW + v];
    float d0 = se0-cx, d1 = se1-cy, d2 = se2-cz;
    float dist = fast_exp(-(d0*d0*e0 + d1*d1*e1 + d2*d2*e2));
    int m = (inst[(long long)b*DHW + v] == iid);
    float err = m ? (2.f - 2.f*dist) : (2.f*dist);
    int bin = (int)(err * (NBINS*0.5f));
    bin = bin < 0 ? 0 : (bin > NBINS-1 ? NBINS-1 : bin);
    if (m) {
      atomicAdd(&lcp[bin], 1u);
      float sd = fast_sigmoid(predB[6*(long long)DHW + v]);
      float ds = sd - dist; sA += ds*ds;
      float t0 = predB[3*(long long)DHW+v]-sm0;
      float t1 = predB[4*(long long)DHW+v]-sm1;
      float t2 = predB[5*(long long)DHW+v]-sm2;
      vA += t0*t0 + t1*t1 + t2*t2;
    } else {
      atomicAdd(&lcn[bin], 1u);
    }
    atomicAdd(&lsum[bin], err);
  }

  for (int o = 16; o > 0; o >>= 1) {
    sA += __shfl_down(sA, o, 32);
    vA += __shfl_down(vA, o, 32);
  }
  if ((tid & 31) == 0) {
    atomicAdd(&seedAcc[b*NIDS + id], sA);
    atomicAdd(&varAcc [b*NIDS + id], vA);
  }
  __syncthreads();

  long long hb = (long long)(b*NIDS + id) * NBINS;
  for (int i = tid; i < NBINS; i += 256) {
    unsigned cp = lcp[i], cn = lcn[i];
    if (cp) atomicAdd(&hcp[hb + i], cp);
    if (cn) atomicAdd(&hcn[hb + i], cn);
    if (cp | cn) atomicAdd(&hsum[hb + i], lsum[i]);
  }
}

// =====================================================================
// Phase 4: histogram Lovász-hinge, parallel per (b,id) block.
//   loss = sum_bins mean_err * (jaccard(incl prefix) - jaccard(excl prefix))
//   where prefixes are over bins in DESCENDING error order.
// =====================================================================
__global__ __launch_bounds__(256) void phase4_lovasz(
    const unsigned* __restrict__ hcp, const unsigned* __restrict__ hcn,
    const float* __restrict__ hsum, float* __restrict__ instL)
{
  __shared__ float scp[NBINS];
  __shared__ float scn[NBINS];
  __shared__ float serr[NBINS];
  __shared__ float tsP[257];
  __shared__ float tsN[256];
  __shared__ float blockAcc[8];

  const int slot = blockIdx.x;                 // b*16 + id
  const int tid  = threadIdx.x;
  const long long hb = (long long)slot * NBINS;

  // stage in reversed (descending-error) order
  for (int i = tid; i < NBINS; i += 256) {
    int src = NBINS - 1 - i;
    scp[i]  = (float)hcp[hb + src];
    scn[i]  = (float)hcn[hb + src];
    serr[i] = hsum[hb + src];
  }
  __syncthreads();

  // per-thread serial inclusive scan over its contiguous 8-bin segment
  const int s0 = tid * (NBINS/256);
  float lp[NBINS/256], ln[NBINS/256];
  float accP = 0.f, accN = 0.f;
#pragma unroll
  for (int k = 0; k < NBINS/256; ++k) {
    accP += scp[s0+k]; lp[k] = accP;
    accN += scn[s0+k]; ln[k] = accN;
  }
  tsP[tid] = accP; tsN[tid] = accN;
  __syncthreads();
  if (tid == 0) {                              // exclusive scan of partials
    float rp = 0.f, rn = 0.f;
    for (int t = 0; t < 256; ++t) {
      float tp = tsP[t], tn = tsN[t];
      tsP[t] = rp; tsN[t] = rn;
      rp += tp; rn += tn;
    }
    tsP[256] = rp;                             // total positives P
  }
  __syncthreads();

  const float P = tsP[256];
  const float offP = tsP[tid], offN = tsN[tid];
  float loss = 0.f;
#pragma unroll
  for (int k = 0; k < NBINS/256; ++k) {
    float cp = scp[s0+k], cn = scn[s0+k];
    float cnt = cp + cn;
    if (cnt > 0.f) {
      float pI = offP + lp[k], nI = offN + ln[k];
      float pE = pI - cp,      nE = nI - cn;
      float dI = P + nI,       dE = P + nE;
      float jI = (dI > 0.f) ? (1.f - (P - pI)/dI) : 0.f;
      float jE = (dE > 0.f) ? (1.f - (P - pE)/dE) : 0.f;
      loss += (serr[s0+k]/cnt) * (jI - jE);
    }
  }
  for (int o = 16; o > 0; o >>= 1) loss += __shfl_down(loss, o, 32);
  if ((tid & 31) == 0) blockAcc[tid >> 5] = loss;
  __syncthreads();
  if (tid == 0) {
    float tot = 0.f;
#pragma unroll
    for (int wv = 0; wv < 8; ++wv) tot += blockAcc[wv];
    instL[slot] = tot;                         // exists-mask applied in phase 5
  }
}

// =====================================================================
// Phase 5: combine into the 4 output losses
// =====================================================================
__global__ void phase5_final(const float* __restrict__ params,
                             const float* __restrict__ instL,
                             const float* __restrict__ seedAcc,
                             const float* __restrict__ varAcc,
                             const float* __restrict__ bg,
                             float* __restrict__ out)
{
  if (threadIdx.x != 0 || blockIdx.x != 0) return;
  float li = 0.f, lv = 0.f, ls = 0.f;
  for (int b = 0; b < BB; ++b) {
    float obj = 0.f, vb = 0.f, ib = 0.f, sb = 0.f;
    for (int id = 0; id < NIDS; ++id) {
      const float* p = params + (b*NIDS + id)*12;
      float ex = p[10], cnt = p[9];
      obj += ex;
      float safe = fmaxf(cnt, 1.f);
      vb += ex * (varAcc[b*NIDS + id] / (3.f * safe));
      ib += ex * instL[b*NIDS + id];
      sb += seedAcc[b*NIDS + id];
    }
    float denom = fmaxf(obj, 1.f);
    li += ib / denom;
    lv += vb / denom;
    ls += (sb + bg[b]) / (float)DHW;
  }
  li *= (1.0f / (float)BB);          // W_INST = 1
  lv *= (10.0f / (float)BB);         // W_VAR  = 10
  ls *= (1.0f / (float)BB);          // W_SEED = 1
  out[0] = li; out[1] = lv; out[2] = ls; out[3] = li + lv + ls;
}

// =====================================================================
extern "C" void kernel_launch(void* const* d_in, const int* in_sizes, int n_in,
                              void* d_out, int out_size, void* d_ws, size_t ws_size,
                              hipStream_t stream)
{
  const float* pred   = (const float*)d_in[0];
  const int*   inst   = (const int*)  d_in[1];
  const int*   labels = (const int*)  d_in[2];
  // d_in[3] = center_images (unused by the reference)
  const float* xyzm   = (const float*)d_in[4];

  float* ws = (float*)d_ws;
  float*    stats   = ws;                 // 4096
  float*    bg      = ws + 4096;          // 4
  float*    seedAcc = ws + 4100;          // 64
  float*    varAcc  = ws + 4164;          // 64
  float*    instL   = ws + 4228;          // 64
  float*    params  = ws + 4292;          // 768
  unsigned* hcp     = (unsigned*)(ws + 5120);          // 64*NBINS
  unsigned* hcn     = hcp + BB*NIDS*NBINS;             // 64*NBINS
  float*    hsum    = (float*)(hcn + BB*NIDS*NBINS);   // 64*NBINS

  size_t zbytes = (size_t)(5120 + 3*BB*NIDS*NBINS) * 4;
  (void)hipMemsetAsync(d_ws, 0, zbytes, stream);

  phase1_stats <<<dim3(DHW/CHUNK1, BB),       256, 0, stream>>>(pred, inst, labels, xyzm, stats, bg);
  phase2_params<<<1,                           64, 0, stream>>>(stats, params);
  phase3_dist  <<<dim3(DHW/CHUNK3, NIDS, BB), 256, 0, stream>>>(pred, inst, xyzm, params,
                                                                hcp, hcn, hsum, seedAcc, varAcc);
  phase4_lovasz<<<BB*NIDS,                    256, 0, stream>>>(hcp, hcn, hsum, instL);
  phase5_final <<<1,                           32, 0, stream>>>(params, instL, seedAcc, varAcc, bg,
                                                                (float*)d_out);
}
